// ScaledDotProduct_51127290692267
// MI455X (gfx1250) — matrix operand order repaired and checked by reading, not compiled, not run on hardware
//
#include <hip/hip_runtime.h>

// SDPA forward (flash-attention) for [B=2, H=16, S=2048, Dh=64] fp32.
// f16 WMMA compute (v_wmma_f32_16x16x32_f16), f32 online softmax,
// K/V chunk staging via TENSOR_LOAD_TO_LDS (TDM) overlapped with compute.

typedef __attribute__((ext_vector_type(16))) _Float16     v16h;
typedef __attribute__((ext_vector_type(8)))  float        v8f;
typedef __attribute__((ext_vector_type(4)))  float        v4f;
typedef __attribute__((ext_vector_type(4)))  _Float16     v4h;
typedef __attribute__((ext_vector_type(4)))  unsigned int v4u;
typedef __attribute__((ext_vector_type(8)))  int          v8i;
typedef __attribute__((ext_vector_type(4)))  int          v4i;

#define BH      32      // B*H
#define SEQ     2048
#define DH      64
#define QTILE   128     // q rows per workgroup (8 waves * 16 rows)
#define KCHUNK  64      // keys per iteration

// ds_swizzle immediate XOR within 32 lanes: and_mask=0x1f, or=0, xor in [14:10]
#define SWZ_XOR(x, m) \
    __int_as_float(__builtin_amdgcn_ds_swizzle(__float_as_int(x), ((m) << 10) | 0x1F))

// A-matrix (16x32 f16) K-index for slot j (0..15) given lane-half hf (0/1).
__device__ __forceinline__ int a_kmap(int j, int hf) {
    return 16 * hf + j;
}

// Issue a 2-D TDM tile load: global (fp32, row stride = stride0 elems) -> LDS.
// D# layout per CDNA5 ISA 8.3/8.4: group0 = {count/type/addr}, group1 = dims.
// 6-arg builtin (clang-23 form): (g0, g1, g2, g3, g_extra, cpol).
__device__ __forceinline__ void tdm_load_2d(unsigned lds_addr, const void* gptr,
                                            unsigned tensor_d0, unsigned tensor_d1,
                                            unsigned stride0,
                                            unsigned tile_d0, unsigned tile_d1)
{
    unsigned long long ga = (unsigned long long)(uintptr_t)gptr;
    v4u g0;
    g0[0] = 1u;                                           // count=1 (valid D#)
    g0[1] = lds_addr;                                     // LDS byte address
    g0[2] = (unsigned)(ga & 0xFFFFFFFFu);                 // global_addr[31:0]
    g0[3] = (unsigned)((ga >> 32) & 0x01FFFFFFu)          // global_addr[56:32]
          | (2u << 30);                                   // type=2 ("image")
    v8i g1;
    g1[0] = (int)(2u << 16);                              // data_size=2 -> 4B
    g1[1] = (int)(tensor_d0 << 16);                       // tensor_dim0[15:0]
    g1[2] = (int)((tensor_d0 >> 16) | (tensor_d1 << 16)); // dim0 hi | dim1 lo
    g1[3] = (int)((tensor_d1 >> 16) | (tile_d0 << 16));   // dim1 hi | tile_dim0
    g1[4] = (int)(tile_d1 & 0xFFFFu);                     // tile_dim1 (dim2=0)
    g1[5] = (int)stride0;                                 // tensor_dim0_stride lo
    g1[6] = 0;                                            // stride0 hi | stride1 lo
    g1[7] = 0;                                            // stride1 hi
    v4i z4 = {};
    v8i z8 = {};
    __builtin_amdgcn_tensor_load_to_lds(g0, g1, z4, z4, z8, 0);
}

__global__ __launch_bounds__(256)
void sdpa_fwd_wmma(const float* __restrict__ Q,
                   const float* __restrict__ K,
                   const float* __restrict__ V,
                   float* __restrict__ O)
{
    __shared__ float    Kf32[KCHUNK][DH];         // TDM landing zone, 16 KB
    __shared__ float    Vf32[KCHUNK][DH];         // TDM landing zone, 16 KB
    __shared__ _Float16 Ksh[KCHUNK][DH + 8];      // [key][d] f16, 9 KB
    __shared__ _Float16 VshT[DH][KCHUNK + 8];     // [d][key] f16 (transposed), 9 KB
    __shared__ _Float16 Psh[8][16][KCHUNK + 2];   // per-wave P, [row][key], 16.5 KB

    const int tid  = threadIdx.x;
    const int lane = tid & 31;
    const int wave = tid >> 5;
    const int hf   = lane >> 4;
    const int l16  = lane & 15;

    const int bh    = blockIdx.y;
    const int qbase = blockIdx.x * QTILE + wave * 16;

    const float* qb = Q + (size_t)bh * SEQ * DH;
    const float* kb = K + (size_t)bh * SEQ * DH;
    const float* vb = V + (size_t)bh * SEQ * DH;
    float*       ob = O + (size_t)bh * SEQ * DH;

    const unsigned ldsK = (unsigned)(uintptr_t)&Kf32[0][0];
    const unsigned ldsV = (unsigned)(uintptr_t)&Vf32[0][0];

    union AF { v16h v; _Float16 h[16]; };
    union CF { v8f  v; float    f[8];  };

    // ---- Q A-fragments, pre-scaled by 1/sqrt(Dh) ----
    AF qa[2];
    {
        const float* qrow = qb + (size_t)(qbase + l16) * DH;
#pragma unroll
        for (int dc = 0; dc < 2; ++dc)
#pragma unroll
            for (int j = 0; j < 16; ++j)
                qa[dc].h[j] = (_Float16)(qrow[dc * 32 + a_kmap(j, hf)] * 0.125f);
    }

    float mrun[8], lrun[8];
    v8f acc[4];
    const v8f vzero = {};
#pragma unroll
    for (int r = 0; r < 8; ++r) { mrun[r] = -1e30f; lrun[r] = 0.0f; }
#pragma unroll
    for (int t = 0; t < 4; ++t) acc[t] = vzero;

    // ---- prologue: kick off TDM for chunk 0 (K and V tiles) ----
    if (wave == 0) {
        tdm_load_2d(ldsK, kb, DH, SEQ, DH, DH, KCHUNK);
        tdm_load_2d(ldsV, vb, DH, SEQ, DH, DH, KCHUNK);
    }

    for (int kc = 0; kc < SEQ; kc += KCHUNK) {
        if (wave == 0)
            __builtin_amdgcn_s_wait_tensorcnt(0);   // chunk data landed in LDS
        __syncthreads();

        // ---- fp32 -> f16 conversion pass (K row-major, V transposed) ----
        {
            int kk  = tid >> 2;              // key row (0..63)
            int d16 = (tid & 3) << 4;        // 16-float slice within row
            const v4f* krow = (const v4f*)&Kf32[kk][d16];
            const v4f* vrow = (const v4f*)&Vf32[kk][d16];
#pragma unroll
            for (int q4 = 0; q4 < 4; ++q4) {
                v4f kq = krow[q4];
                v4f vq = vrow[q4];
                v4h kh = { (_Float16)kq[0], (_Float16)kq[1],
                           (_Float16)kq[2], (_Float16)kq[3] };
                *(v4h*)&Ksh[kk][d16 + q4 * 4] = kh;
                int d = d16 + q4 * 4;
                VshT[d + 0][kk] = (_Float16)vq[0];
                VshT[d + 1][kk] = (_Float16)vq[1];
                VshT[d + 2][kk] = (_Float16)vq[2];
                VshT[d + 3][kk] = (_Float16)vq[3];
            }
        }
        __syncthreads();

        // ---- landing zone free: issue next chunk's TDM, overlapped with compute ----
        if (wave == 0 && kc + KCHUNK < SEQ) {
            tdm_load_2d(ldsK, kb + (size_t)(kc + KCHUNK) * DH, DH, SEQ, DH, DH, KCHUNK);
            tdm_load_2d(ldsV, vb + (size_t)(kc + KCHUNK) * DH, DH, SEQ, DH, DH, KCHUNK);
        }

        // ---- scores: S = (Q/sqrt(d)) * K^T, four 16-key tiles ----
        CF s[4];
#pragma unroll
        for (int t = 0; t < 4; ++t) {
            s[t].v = vzero;
#pragma unroll
            for (int dc = 0; dc < 2; ++dc) {
                AF kf;   // B 32x16: col N=l16 (key t*16+l16), K = dc*32+16*hf+j
#pragma unroll
                for (int j = 0; j < 16; ++j)
                    kf.h[j] = Ksh[t * 16 + l16][dc * 32 + 16 * hf + j];
                s[t].v = __builtin_amdgcn_wmma_f32_16x16x32_f16(
                    false, qa[dc].v, false, kf.v, (short)0, s[t].v, false, false);
            }
        }

        // ---- online softmax (row = r + 8*hf; reduce across 16-lane half) ----
#pragma unroll
        for (int r = 0; r < 8; ++r) {
            float mx = fmaxf(fmaxf(s[0].f[r], s[1].f[r]),
                             fmaxf(s[2].f[r], s[3].f[r]));
            mx = fmaxf(mx, SWZ_XOR(mx, 8));
            mx = fmaxf(mx, SWZ_XOR(mx, 4));
            mx = fmaxf(mx, SWZ_XOR(mx, 2));
            mx = fmaxf(mx, SWZ_XOR(mx, 1));
            float mnew = fmaxf(mrun[r], mx);
            float corr = __expf(mrun[r] - mnew);
            mrun[r] = mnew;
            float rs = 0.0f;
#pragma unroll
            for (int t = 0; t < 4; ++t) {
                float p = __expf(s[t].f[r] - mnew);
                s[t].f[r] = p;
                rs += p;
            }
            rs += SWZ_XOR(rs, 8);
            rs += SWZ_XOR(rs, 4);
            rs += SWZ_XOR(rs, 2);
            rs += SWZ_XOR(rs, 1);
            lrun[r] = lrun[r] * corr + rs;
#pragma unroll
            for (int t = 0; t < 4; ++t) acc[t][r] *= corr;
        }

        // ---- transpose P through LDS (same wave, DS in-order) ----
#pragma unroll
        for (int r = 0; r < 8; ++r)
#pragma unroll
            for (int t = 0; t < 4; ++t)
                Psh[wave][r + 8 * hf][t * 16 + l16] = (_Float16)s[t].f[r];

        AF pa[2];
#pragma unroll
        for (int kc2 = 0; kc2 < 2; ++kc2)
#pragma unroll
            for (int j = 0; j < 16; ++j)
                pa[kc2].h[j] = Psh[wave][l16][kc2 * 32 + a_kmap(j, hf)];

        // ---- acc += P * V : 4 Dh-tiles x 2 key-subchunks (K=32 each) ----
#pragma unroll
        for (int t2 = 0; t2 < 4; ++t2) {
#pragma unroll
            for (int kc2 = 0; kc2 < 2; ++kc2) {
                AF vf;   // contiguous 16 halves from transposed V
#pragma unroll
                for (int j = 0; j < 16; ++j)
                    vf.h[j] = VshT[t2 * 16 + l16][kc2 * 32 + 16 * hf + j];
                acc[t2] = __builtin_amdgcn_wmma_f32_16x16x32_f16(
                    false, pa[kc2].v, false, vf.v, (short)0, acc[t2], false, false);
            }
        }
        // loop's top barrier (after tensor-wait) fences Ksh/VshT reuse
    }

    // ---- epilogue: O = acc / l ----
#pragma unroll
    for (int r = 0; r < 8; ++r) {
        float inv = 1.0f / lrun[r];
        int row = qbase + r + 8 * hf;
#pragma unroll
        for (int t = 0; t < 4; ++t)
            ob[(size_t)row * DH + t * 16 + l16] = acc[t][r] * inv;
    }
}

extern "C" void kernel_launch(void* const* d_in, const int* in_sizes, int n_in,
                              void* d_out, int out_size, void* d_ws, size_t ws_size,
                              hipStream_t stream) {
    const float* q = (const float*)d_in[0];
    const float* k = (const float*)d_in[1];
    const float* v = (const float*)d_in[2];
    float* out = (float*)d_out;

    dim3 grid(SEQ / QTILE, BH);   // (16, 32)
    dim3 block(256);              // 8 waves
    sdpa_fwd_wmma<<<grid, block, 0, stream>>>(q, k, v, out);
}
